// LanguageModelOneHot_2808908612119
// MI455X (gfx1250) — compile-verified, matrix operand order checked by reading, not run on hardware
//
#include <hip/hip_runtime.h>
#include <cstdint>

// ---------------------------------------------------------------------------
// 2-layer LSTM LM on gfx1250 (MI455X): bf16 WMMA 16x16x32 w/ f32 accum.
//  - 512 fused LSTM step kernels (latency-bound recurrence, graph-replayed)
//  - one blocked GEMM kernel (LDS-staged, async-to-LDS double buffered) used
//    for both the inter-layer projection and the 134-GFLOP final FC.
// ---------------------------------------------------------------------------

typedef __attribute__((ext_vector_type(16))) __bf16 v16bf;
typedef __attribute__((ext_vector_type(8)))  __bf16 v8bf;
typedef __attribute__((ext_vector_type(8)))  float  v8f;
typedef __attribute__((ext_vector_type(4)))  float  v4f;
typedef __attribute__((ext_vector_type(4)))  int    v4i;

#define HDIM 1024   // hidden
#define GDIM 4096   // 4*H gate width
#define SEQ  256
#define BATCH 8
#define VOC  32000

#define NB_TILES 8   // 8 x 16 = 128 output cols per WG
#define SB_TILES 16  // 16 s-tiles (16 padded rows each) per WG

#define WMMA_BF16(a, b, c) \
    __builtin_amdgcn_wmma_f32_16x16x32_bf16(false, (a), false, (b), (short)0, (c), false, false)

#if defined(__has_builtin)
#if __has_builtin(__builtin_amdgcn_global_load_async_to_lds_b128)
#define ASYNC_LDS 1
#endif
#endif

__device__ __forceinline__ float sigmoid_f(float x) { return 1.0f / (1.0f + __expf(-x)); }

// Load one 16-element bf16 WMMA operand from a row-major [*][row_stride]
// buffer.  ISA 16-bit operand layout:
//   lanes 0-15  hold K[k..k+7]  and K[k+16..k+23]
//   lanes 16-31 hold K[k+8..k+15] and K[k+24..k+31]
__device__ __forceinline__ v16bf load_op_bf16(const __bf16* __restrict__ base,
                                              int row_stride, int row, int k, int lane) {
    int koff = (lane >> 4) << 3;                       // 0 or 8
    const __bf16* p = base + (size_t)row * row_stride + k + koff;
    v8bf lo = *(const v8bf*)(p);
    v8bf hi = *(const v8bf*)(p + 16);
    v16bf r;
#pragma unroll
    for (int j = 0; j < 8; ++j) { r[j] = lo[j]; r[8 + j] = hi[j]; }
    return r;
}

// Operand read from an LDS-resident [16][32] bf16 tile (one 32-wide K chunk).
__device__ __forceinline__ v16bf lds_op(const __bf16* tile, int nl, int lane) {
    int koff = (lane >> 4) << 3;
    const __bf16* p = tile + nl * 32 + koff;
    v8bf lo = *(const v8bf*)(p);
    v8bf hi = *(const v8bf*)(p + 16);
    v16bf r;
#pragma unroll
    for (int j = 0; j < 8; ++j) { r[j] = lo[j]; r[8 + j] = hi[j]; }
    return r;
}

__device__ __forceinline__ void wait_async0() {
#if defined(ASYNC_LDS)
#if __has_builtin(__builtin_amdgcn_s_wait_asynccnt)
    __builtin_amdgcn_s_wait_asynccnt(0);
#else
    asm volatile("s_wait_asynccnt 0" ::: "memory");
#endif
#endif
}

// --------------------------------------------------------------------------
// 1) Embedding gather: xp0[s][b][j] = w_ih0[j, x[b,s]] + b_ih0[j] + b_hh0[j]
// --------------------------------------------------------------------------
__global__ __launch_bounds__(256) void embed_kernel(
    const int* __restrict__ x, const float* __restrict__ w_ih0,
    const float* __restrict__ b_ih, const float* __restrict__ b_hh,
    float* __restrict__ xp0) {
    int idx = blockIdx.x;                 // b*SEQ + s
    int b = idx / SEQ, s = idx % SEQ;
    int tok = x[idx];
    float* out = xp0 + ((size_t)s * BATCH + b) * GDIM;
    for (int j = threadIdx.x; j < GDIM; j += 256)
        out[j] = w_ih0[(size_t)j * VOC + tok] + b_ih[j] + b_hh[j];
}

// --------------------------------------------------------------------------
// 2) f32 -> bf16 conversion (recurrent weights, read 256x -> convert once)
// --------------------------------------------------------------------------
__global__ __launch_bounds__(256) void cvt_bf16_kernel(
    const float* __restrict__ src, __bf16* __restrict__ dst, int n) {
    int i = blockIdx.x * 256 + threadIdx.x;
    if (i < n) dst[i] = (__bf16)src[i];
}

// --------------------------------------------------------------------------
// 3) bias1[i] = b_ih1[i] + b_hh1[i]
// --------------------------------------------------------------------------
__global__ __launch_bounds__(256) void bias_sum_kernel(
    const float* __restrict__ a, const float* __restrict__ b,
    float* __restrict__ dst, int n) {
    int i = blockIdx.x * 256 + threadIdx.x;
    if (i < n) dst[i] = a[i] + b[i];
}

// --------------------------------------------------------------------------
// 4) Initialize padded h tiles + c/h f32 states from h0/c0 inputs
// --------------------------------------------------------------------------
__global__ __launch_bounds__(256) void init_states_kernel(
    const float* __restrict__ h0, const float* __restrict__ c0,
    __bf16* __restrict__ hinit0, __bf16* __restrict__ hinit1,
    float* __restrict__ cst0, float* __restrict__ cst1,
    float* __restrict__ hst0, float* __restrict__ hst1) {
    int i = blockIdx.x * 256 + threadIdx.x;     // over 16*HDIM padded tile
    if (i >= 16 * HDIM) return;
    int m = i >> 10, k = i & (HDIM - 1);
    float v0 = (m < BATCH) ? h0[0 * BATCH * HDIM + m * HDIM + k] : 0.0f;
    float v1 = (m < BATCH) ? h0[1 * BATCH * HDIM + m * HDIM + k] : 0.0f;
    hinit0[i] = (__bf16)v0;
    hinit1[i] = (__bf16)v1;
    if (m < BATCH) {
        cst0[m * HDIM + k] = c0[0 * BATCH * HDIM + m * HDIM + k];
        cst1[m * HDIM + k] = c0[1 * BATCH * HDIM + m * HDIM + k];
        hst0[m * HDIM + k] = v0;
        hst1[m * HDIM + k] = v1;
    }
}

// --------------------------------------------------------------------------
// 5) One fused LSTM time step: g = xp[s] + h_prev @ w_hh^T, then gates.
//    Each wave owns one 16-column slice of H and all four gates of it.
//    Grid: 8 blocks x 8 waves = 64 tiles = H/16.  Latency-bound; codegen
//    verified: b128 loads software-pipelined against back-to-back WMMAs.
// --------------------------------------------------------------------------
__global__ __launch_bounds__(256) void lstm_step_kernel(
    const float* __restrict__ xp_s,      // [BATCH][GDIM] for this step
    const __bf16* __restrict__ whh,      // [GDIM][HDIM] bf16
    const __bf16* __restrict__ h_in,     // [16][HDIM] padded prev h
    __bf16* __restrict__ h_out,          // [16][HDIM] padded new h (= hs[s])
    float* __restrict__ c_state,         // [BATCH][HDIM]
    float* __restrict__ h_state) {       // [BATCH][HDIM] f32 (for hn output)
    int wave = threadIdx.x >> 5;
    int lane = threadIdx.x & 31;
    int tile = blockIdx.x * 8 + wave;    // 0..63
    int n0 = tile * 16;
    int nl = lane & 15;
    int col = n0 + nl;
    bool lo = lane < 16;

    v8f acc[4];
#pragma unroll
    for (int g = 0; g < 4; ++g)
#pragma unroll
        for (int v = 0; v < 8; ++v)
            acc[g][v] = lo ? xp_s[v * GDIM + g * HDIM + col] : 0.0f;

    for (int k = 0; k < HDIM; k += 32) {
        v16bf a = load_op_bf16(h_in, HDIM, nl, k, lane);
#pragma unroll
        for (int g = 0; g < 4; ++g) {
            v16bf b = load_op_bf16(whh, HDIM, g * HDIM + col, k, lane);
            acc[g] = WMMA_BF16(a, b, acc[g]);
        }
    }

    if (lo) {
#pragma unroll
        for (int v = 0; v < 8; ++v) {    // m = v = batch row (0..7 all valid)
            float iv = sigmoid_f(acc[0][v]);
            float fv = sigmoid_f(acc[1][v]);
            float gv = tanhf(acc[2][v]);
            float ov = sigmoid_f(acc[3][v]);
            float cold = c_state[v * HDIM + col];
            float cnew = fv * cold + iv * gv;
            float hnew = ov * tanhf(cnew);
            c_state[v * HDIM + col] = cnew;
            h_state[v * HDIM + col] = hnew;
            h_out[v * HDIM + col] = (__bf16)hnew;
        }
    } else {
#pragma unroll
        for (int v = 0; v < 8; ++v)       // rows 8..15: keep padding zero
            h_out[(v + 8) * HDIM + col] = (__bf16)0.0f;
    }
}

// --------------------------------------------------------------------------
// 6) Blocked GEMM with LDS staging (used for xp1 projection and final FC):
//      out[s, m, col] = sum_k A[s][m][k] * B[col][k] + bias[col]
//    WG tile: 128 cols (8 n-tiles) x 16 s-tiles; wave: 2 s-tiles x 8 n-tiles
//    = 16 WMMA accumulators.  Per 32-K chunk:
//      - A panel (16 KB bf16) -> LDS via global_load_async_to_lds_b128
//      - B panel (f32) -> cvt bf16 -> ds_store (halves compute-side LDS reads)
//    Double-buffered, s_wait_asynccnt + one barrier per chunk.
// --------------------------------------------------------------------------
__device__ __forceinline__ void stage_A(const __bf16* __restrict__ A,
                                        int s_tile0, int k,
                                        __bf16* __restrict__ dst, int tid) {
#pragma unroll
    for (int j = 0; j < 4; ++j) {
        int li = j * 256 + tid;          // 0..1023, 16 B each = 16 KB
        int st  = li >> 6;               // s-tile 0..15
        int row = (li >> 2) & 15;        // padded row
        int sub = li & 3;                // 8-elem group within 32-K chunk
        const __bf16* g = A + ((size_t)(s_tile0 + st) * 16 + row) * HDIM + k + sub * 8;
        __bf16* l = dst + (st * 16 + row) * 32 + sub * 8;
#if defined(ASYNC_LDS)
        __builtin_amdgcn_global_load_async_to_lds_b128(
            (__attribute__((address_space(1))) v4i*)(uintptr_t)g,
            (__attribute__((address_space(3))) v4i*)(v4i*)l, 0, 0);
#else
        *(v8bf*)l = *(const v8bf*)g;
#endif
    }
}

__device__ __forceinline__ void stage_B(const float* __restrict__ B,
                                        int n0, int k,
                                        __bf16* __restrict__ dst, int tid) {
    int row  = tid >> 1;                 // 0..127
    int half = tid & 1;                  // 16-f32 half of the 32-K chunk
    const float* g = B + (size_t)(n0 + row) * HDIM + k + half * 16;
    __bf16* l = dst + row * 32 + half * 16;
#pragma unroll
    for (int j = 0; j < 4; ++j) {
        v4f v = *(const v4f*)(g + j * 4);
#pragma unroll
        for (int q = 0; q < 4; ++q) l[j * 4 + q] = (__bf16)v[q];
    }
}

__global__ __launch_bounds__(256) void gemm_blocked_kernel(
    const __bf16* __restrict__ A,        // [SEQ][16][HDIM] padded bf16
    const float* __restrict__ B,         // [N][HDIM] f32 (read once -> cvt)
    const float* __restrict__ bias,      // [N]
    float* __restrict__ out,
    long s_stride, long m_stride) {      // out + s*s_stride + m*m_stride + col
    __shared__ __bf16 lA[2][SB_TILES * 16 * 32];   // 2 x 16 KB
    __shared__ __bf16 lB[2][NB_TILES * 16 * 32];   // 2 x  8 KB

    int tid  = threadIdx.x;
    int wave = tid >> 5;
    int lane = tid & 31;
    int nl   = lane & 15;
    int n0      = blockIdx.x * (NB_TILES * 16);
    int s_tile0 = blockIdx.y * SB_TILES;

    v8f acc[2][NB_TILES];
#pragma unroll
    for (int si = 0; si < 2; ++si)
#pragma unroll
        for (int nt = 0; nt < NB_TILES; ++nt) {
            float bv = bias[n0 + nt * 16 + nl];
#pragma unroll
            for (int v = 0; v < 8; ++v) acc[si][nt][v] = bv;
        }

    int buf = 0;
    stage_A(A, s_tile0, 0, &lA[0][0], tid);
    stage_B(B, n0, 0, &lB[0][0], tid);

    for (int kc = 0; kc < HDIM / 32; ++kc) {
        wait_async0();                   // own async A copies done
        __syncthreads();                 // panel kc visible to all waves
        if (kc + 1 < HDIM / 32) {        // prefetch next chunk into other buf
            stage_A(A, s_tile0, (kc + 1) * 32, &lA[buf ^ 1][0], tid);
            stage_B(B, n0, (kc + 1) * 32, &lB[buf ^ 1][0], tid);
        }
        v16bf a0 = lds_op(&lA[buf][(wave * 2 + 0) * 16 * 32], nl, lane);
        v16bf a1 = lds_op(&lA[buf][(wave * 2 + 1) * 16 * 32], nl, lane);
#pragma unroll
        for (int nt = 0; nt < NB_TILES; ++nt) {
            v16bf b = lds_op(&lB[buf][nt * 16 * 32], nl, lane);
            acc[0][nt] = WMMA_BF16(a0, b, acc[0][nt]);
            acc[1][nt] = WMMA_BF16(a1, b, acc[1][nt]);
        }
        buf ^= 1;
    }

    if (lane < 16) {
#pragma unroll
        for (int si = 0; si < 2; ++si) {
            int s = s_tile0 + wave * 2 + si;      // global s (tile == step)
#pragma unroll
            for (int nt = 0; nt < NB_TILES; ++nt) {
                int col = n0 + nt * 16 + nl;
#pragma unroll
                for (int v = 0; v < 8; ++v)       // m = v = batch row
                    out[(size_t)s * s_stride + (size_t)v * m_stride + col] =
                        acc[si][nt][v];
            }
        }
    }
}

// --------------------------------------------------------------------------
// 7) Emit hn/cn after the logits block of d_out
// --------------------------------------------------------------------------
__global__ __launch_bounds__(256) void write_states_kernel(
    const float* __restrict__ h1, const float* __restrict__ h2,
    const float* __restrict__ c1, const float* __restrict__ c2,
    float* __restrict__ out) {           // points at d_out + B*S*V
    int i = blockIdx.x * 256 + threadIdx.x;   // over BATCH*HDIM
    if (i >= BATCH * HDIM) return;
    out[i]                      = h1[i];
    out[BATCH * HDIM + i]       = h2[i];
    out[2 * BATCH * HDIM + i]   = c1[i];
    out[3 * BATCH * HDIM + i]   = c2[i];
}

// ---------------------------------------------------------------------------
extern "C" void kernel_launch(void* const* d_in, const int* in_sizes, int n_in,
                              void* d_out, int out_size, void* d_ws, size_t ws_size,
                              hipStream_t stream) {
    (void)in_sizes; (void)n_in; (void)out_size; (void)ws_size;

    const int*   x     = (const int*)  d_in[0];
    const float* h0    = (const float*)d_in[1];
    const float* c0    = (const float*)d_in[2];
    const float* w_ih0 = (const float*)d_in[3];
    const float* w_hh0 = (const float*)d_in[4];
    const float* b_ih0 = (const float*)d_in[5];
    const float* b_hh0 = (const float*)d_in[6];
    const float* w_ih1 = (const float*)d_in[7];
    const float* w_hh1 = (const float*)d_in[8];
    const float* b_ih1 = (const float*)d_in[9];
    const float* b_hh1 = (const float*)d_in[10];
    const float* fc_w  = (const float*)d_in[11];
    const float* fc_b  = (const float*)d_in[12];
    float* out = (float*)d_out;

    // ---- workspace carve-up (~101 MB) ----
    char* ws = (char*)d_ws;
    size_t off = 0;
    auto carve = [&](size_t bytes) -> void* {
        void* p = ws + off;
        off = (off + bytes + 255) & ~(size_t)255;
        return p;
    };
    float*  xp0    = (float*) carve((size_t)SEQ * BATCH * GDIM * 4);
    float*  xp1    = (float*) carve((size_t)SEQ * BATCH * GDIM * 4);
    __bf16* whh0b  = (__bf16*)carve((size_t)GDIM * HDIM * 2);
    __bf16* whh1b  = (__bf16*)carve((size_t)GDIM * HDIM * 2);
    __bf16* h1s    = (__bf16*)carve((size_t)SEQ * 16 * HDIM * 2);
    __bf16* h2s    = (__bf16*)carve((size_t)SEQ * 16 * HDIM * 2);
    __bf16* hinit0 = (__bf16*)carve((size_t)16 * HDIM * 2);
    __bf16* hinit1 = (__bf16*)carve((size_t)16 * HDIM * 2);
    float*  cst0   = (float*) carve((size_t)BATCH * HDIM * 4);
    float*  cst1   = (float*) carve((size_t)BATCH * HDIM * 4);
    float*  hst0   = (float*) carve((size_t)BATCH * HDIM * 4);
    float*  hst1   = (float*) carve((size_t)BATCH * HDIM * 4);
    float*  bias1  = (float*) carve((size_t)GDIM * 4);

    // 1) embedding gather + folded biases
    embed_kernel<<<BATCH * SEQ, 256, 0, stream>>>(x, w_ih0, b_ih0, b_hh0, xp0);

    // 2) recurrent weights -> bf16 (read 256x later, mostly from L2)
    cvt_bf16_kernel<<<(GDIM * HDIM) / 256, 256, 0, stream>>>(w_hh0, whh0b, GDIM * HDIM);
    cvt_bf16_kernel<<<(GDIM * HDIM) / 256, 256, 0, stream>>>(w_hh1, whh1b, GDIM * HDIM);
    bias_sum_kernel<<<GDIM / 256, 256, 0, stream>>>(b_ih1, b_hh1, bias1, GDIM);

    // 3) initial states (padded-to-16 bf16 h tiles + f32 c/h)
    init_states_kernel<<<(16 * HDIM + 255) / 256, 256, 0, stream>>>(
        h0, c0, hinit0, hinit1, cst0, cst1, hst0, hst1);

    // 4) LSTM layer 0: 256 sequential fused-step kernels
    for (int s = 0; s < SEQ; ++s) {
        const __bf16* hin = (s == 0) ? hinit0 : (h1s + (size_t)(s - 1) * 16 * HDIM);
        lstm_step_kernel<<<8, 256, 0, stream>>>(
            xp0 + (size_t)s * BATCH * GDIM, whh0b, hin,
            h1s + (size_t)s * 16 * HDIM, cst0, hst0);
    }

    // 5) inter-layer projection for all timesteps at once
    //    out[s][m][col] -> s_stride = BATCH*GDIM, m_stride = GDIM
    gemm_blocked_kernel<<<dim3(GDIM / 128, SEQ / SB_TILES), 256, 0, stream>>>(
        h1s, w_ih1, bias1, xp1, (long)BATCH * GDIM, (long)GDIM);

    // 6) LSTM layer 1
    for (int s = 0; s < SEQ; ++s) {
        const __bf16* hin = (s == 0) ? hinit1 : (h2s + (size_t)(s - 1) * 16 * HDIM);
        lstm_step_kernel<<<8, 256, 0, stream>>>(
            xp1 + (size_t)s * BATCH * GDIM, whh1b, hin,
            h2s + (size_t)s * 16 * HDIM, cst1, hst1);
    }

    // 7) final FC -> logits: row = b*SEQ + s -> s_stride = VOC, m_stride = SEQ*VOC
    gemm_blocked_kernel<<<dim3(VOC / 128, SEQ / SB_TILES), 256, 0, stream>>>(
        h2s, fc_w, fc_b, out, (long)VOC, (long)SEQ * VOC);

    // 8) hn / cn
    write_states_kernel<<<(BATCH * HDIM + 255) / 256, 256, 0, stream>>>(
        hst0, hst1, cst0, cst1, out + (size_t)BATCH * SEQ * VOC);
}
